// ART_16638703305338
// MI455X (gfx1250) — matrix operand order repaired and checked by reading, not compiled, not run on hardware
//
#include <hip/hip_runtime.h>
#include <hip/hip_bf16.h>
#include <math.h>

#define KIMG 8
#define HH 128
#define WW 128
#define NN (HH * WW)   // 16384 tokens per image
#define DD 768
#define CC 10
#define CPAD 16        // pad clusters to 16 for WMMA N-dim
#define KM_ITERS 5
#define TPB 512        // tokens per assign block
#define NCHUNK (DD / 32)  // 24 K-chunks per row
#define PFD 4             // A-fragment prefetch depth

typedef __attribute__((ext_vector_type(16))) __bf16 v16bf;
typedef __attribute__((ext_vector_type(8)))  __bf16 v8bf;
typedef __attribute__((ext_vector_type(8)))  float  v8f;

static __device__ __forceinline__ unsigned short f2bf(float f) {
  union { float f; unsigned u; } v; v.f = f;
  unsigned u = v.u;
  unsigned r = u + 0x7FFFu + ((u >> 16) & 1u);  // round-to-nearest-even
  return (unsigned short)(r >> 16);
}

static __device__ __forceinline__ float gelu_exact(float x) {
  return 0.5f * x * (1.0f + erff(x * 0.70710678118654752440f));
}

// ---------------------------------------------------------------------------
// Kernel 1: tokens = (feat + pos@Wp + bp) * binarize(mask), stored bf16.
// One block per token, 192 threads x float4 (768 dims).
// ---------------------------------------------------------------------------
__global__ void __launch_bounds__(192) build_tokens_k(
    const float* __restrict__ feat, const float* __restrict__ mask,
    const float* __restrict__ boxes, const float* __restrict__ Wp,
    const float* __restrict__ bp, const int* __restrict__ rawh,
    const int* __restrict__ raww, unsigned short* __restrict__ tokens) {
  int token = blockIdx.x;               // k*NN + y*WW + x
  int k = token / NN;
  int rem = token - k * NN;
  int y = rem / WW, x = rem - y * WW;
  float top = boxes[k * 4 + 0], left = boxes[k * 4 + 1];
  float bottom = boxes[k * 4 + 2], right = boxes[k * 4 + 3];
  float rw = (float)(raww[0] - 1), rh = (float)(rawh[0] - 1);
  float xg = ((float)x / (float)WW * (right - left) + left) / rw;
  float yg = ((float)y / (float)HH * (bottom - top) + top) / rh;
  xg = fminf(fmaxf(xg, 0.0f), 1.0f);
  yg = fminf(fmaxf(yg, 0.0f), 1.0f);
  float m = mask[token] > 0.0f ? 1.0f : 0.0f;

  int d = threadIdx.x * 4;
  float4 f4  = *(const float4*)(feat + (size_t)token * DD + d);
  float4 wp0 = *(const float4*)(Wp + d);
  float4 wp1 = *(const float4*)(Wp + DD + d);
  float4 b4  = *(const float4*)(bp + d);
  ushort4 o;
  o.x = f2bf((f4.x + xg * wp0.x + yg * wp1.x + b4.x) * m);
  o.y = f2bf((f4.y + xg * wp0.y + yg * wp1.y + b4.y) * m);
  o.z = f2bf((f4.z + xg * wp0.z + yg * wp1.z + b4.z) * m);
  o.w = f2bf((f4.w + xg * wp0.w + yg * wp1.w + b4.w) * m);
  *(ushort4*)(tokens + (size_t)token * DD + d) = o;
}

// ---------------------------------------------------------------------------
// Kernel 2: init centroids from init_idx (recompute token in f32), compute
// ||c||^2, zero accumulators, set padded rows (c>=10) to zero / +inf norm.
// One block per (k, c) with c in [0,16).
// ---------------------------------------------------------------------------
__global__ void __launch_bounds__(256) init_cent_k(
    const float* __restrict__ feat, const float* __restrict__ boxes,
    const float* __restrict__ Wp, const float* __restrict__ bp,
    const int* __restrict__ init_idx, const int* __restrict__ rawh,
    const int* __restrict__ raww, float* __restrict__ centf,
    unsigned short* __restrict__ centb, float* __restrict__ cn2,
    float* __restrict__ gsums, float* __restrict__ gcnt) {
  int k = blockIdx.x / CPAD, c = blockIdx.x % CPAD;
  size_t row = ((size_t)k * CPAD + c) * DD;
  __shared__ float red[256];
  float ss = 0.0f;
  if (c < CC) {
    int idx = init_idx[k * CC + c];
    int y = idx / WW, x = idx - y * WW;
    float top = boxes[k * 4 + 0], left = boxes[k * 4 + 1];
    float bottom = boxes[k * 4 + 2], right = boxes[k * 4 + 3];
    float rw = (float)(raww[0] - 1), rh = (float)(rawh[0] - 1);
    float xg = ((float)x / (float)WW * (right - left) + left) / rw;
    float yg = ((float)y / (float)HH * (bottom - top) + top) / rh;
    xg = fminf(fmaxf(xg, 0.0f), 1.0f);
    yg = fminf(fmaxf(yg, 0.0f), 1.0f);
    const float* frow = feat + ((size_t)k * NN + idx) * DD;
    for (int jj = 0; jj < 3; ++jj) {
      int d = threadIdx.x + jj * 256;
      float v = frow[d] + xg * Wp[d] + yg * Wp[DD + d] + bp[d];
      centf[row + d] = v;
      centb[row + d] = f2bf(v);
      ss += v * v;
      gsums[row + d] = 0.0f;
    }
  } else {
    for (int jj = 0; jj < 3; ++jj) {
      int d = threadIdx.x + jj * 256;
      centf[row + d] = 0.0f;
      centb[row + d] = 0;
      gsums[row + d] = 0.0f;
    }
  }
  red[threadIdx.x] = ss;
  __syncthreads();
  for (int s = 128; s > 0; s >>= 1) {
    if (threadIdx.x < s) red[threadIdx.x] += red[threadIdx.x + s];
    __syncthreads();
  }
  if (threadIdx.x == 0) {
    cn2[k * CPAD + c] = (c < CC) ? red[0] : 3.0e38f;
    gcnt[k * CPAD + c] = 0.0f;
  }
}

// ---------------------------------------------------------------------------
// Kernel 3: k-means assignment + accumulation.
// One wave per 16-token tile: dot(token, centroid) via v_wmma_f32_16x16x32_bf16
// over 24 K-chunks. Centroids staged once per block into LDS (24 KB), so B
// fragments come from ds_load_b128 (frees ~190 VGPRs -> high occupancy).
// A fragments use a depth-4 rotating prefetch enforced by sched_barrier so 8
// global_load_b128 stay in flight per wave while WMMAs execute.
// argmin(||c||^2 - 2*dot) via shfl_xor within a 16-lane half; masked
// scatter-add into LDS cluster sums; global f32 atomic flush.
// ---------------------------------------------------------------------------
__global__ void __launch_bounds__(256) assign_k(
    const __bf16* __restrict__ tokens, const __bf16* __restrict__ centb,
    const float* __restrict__ cn2, const float* __restrict__ mask,
    float* __restrict__ gsums, float* __restrict__ gcnt) {
  __shared__ float lsum[CC * DD];            // 30720 B
  __shared__ float lcnt[CC];
  __shared__ unsigned short lcentb[CPAD * DD];  // 24576 B centroid stage

  const int segs = NN / TPB;  // 32
  int k = blockIdx.x / segs;
  int seg = blockIdx.x % segs;
  int lane = threadIdx.x & 31;
  int wave = threadIdx.x >> 5;
  int half = lane >> 4;
  int l15 = lane & 15;

  // Zero LDS accumulators and stage this image's centroids into LDS.
  for (int i = threadIdx.x; i < CC * DD; i += 256) lsum[i] = 0.0f;
  if (threadIdx.x < CC) lcnt[threadIdx.x] = 0.0f;
  {
    const uint4* src = (const uint4*)(centb + (size_t)k * CPAD * DD);
    uint4* dst = (uint4*)lcentb;
    for (int i = threadIdx.x; i < CPAD * DD * 2 / 16; i += 256) dst[i] = src[i];
  }
  __syncthreads();

  const __bf16* tok = tokens + (size_t)k * NN * DD;
  const float* mp = mask + (size_t)k * NN;
  float myc2 = cn2[k * CPAD + l15];                 // +inf for padded c>=10
  // B fragment source in LDS: column = centroid l15, K = k0 + half*8 + [0..7], k0 + 16 + half*8 + [0..7]
  const __bf16* brow = (const __bf16*)lcentb + (size_t)l15 * DD + half * 8;

  for (int t = wave; t < TPB / 16; t += 8) {
    int base = seg * TPB + t * 16;
    // A fragment: lane loads token row (base + l15); K-interleave per ISA layout
    const __bf16* arow = tok + (size_t)(base + l15) * DD + half * 8;
    v8f acc = {0.f, 0.f, 0.f, 0.f, 0.f, 0.f, 0.f, 0.f};
    // Depth-PFD rotating prefetch of A fragments.
    v8bf pr0[PFD], pr1[PFD];
#pragma unroll
    for (int i = 0; i < PFD; ++i) {
      pr0[i] = *(const v8bf*)(arow + i * 32);
      pr1[i] = *(const v8bf*)(arow + i * 32 + 16);
    }
#pragma unroll
    for (int kk = 0; kk < NCHUNK; ++kk) {
      int slot = kk % PFD;
      v8bf c0 = pr0[slot], c1 = pr1[slot];
      if (kk < NCHUNK - PFD) {
        int nx = (kk + PFD) * 32;
        pr0[slot] = *(const v8bf*)(arow + nx);
        pr1[slot] = *(const v8bf*)(arow + nx + 16);
      }
      // B fragment from LDS (ds_load_b128 x2, short latency).
      v8bf b0 = *(const v8bf*)(brow + kk * 32);
      v8bf b1 = *(const v8bf*)(brow + kk * 32 + 16);
      // Fence: prefetch (and B loads) must be issued before this point; the
      // WMMA below consumes data loaded PFD iterations ago.
      __builtin_amdgcn_sched_barrier(0);
      v16bf a = __builtin_shufflevector(c0, c1, 0, 1, 2, 3, 4, 5, 6, 7, 8, 9,
                                        10, 11, 12, 13, 14, 15);
      v16bf b = __builtin_shufflevector(b0, b1, 0, 1, 2, 3, 4, 5, 6, 7, 8, 9,
                                        10, 11, 12, 13, 14, 15);
      acc = __builtin_amdgcn_wmma_f32_16x16x32_bf16(false, a, false, b,
                                                    (short)0, acc, false, false);
    }
    // argmin over centroids: element r holds dot(token base+half*8+r, cent l15)
    int labels[8];
#pragma unroll
    for (int r = 0; r < 8; ++r) {
      float v = (l15 < CC) ? (myc2 - 2.0f * acc[r]) : 3.0e38f;
      int idx = l15;
#pragma unroll
      for (int msk = 1; msk <= 8; msk <<= 1) {  // stays within 16-lane half
        float ov = __shfl_xor(v, msk, 32);
        int oi = __shfl_xor(idx, msk, 32);
        if (ov < v || (ov == v && oi < idx)) { v = ov; idx = oi; }
      }
      labels[r] = idx;  // broadcast to all lanes of the half
    }
    // masked scatter-add: half cooperatively adds its 8 tokens (48 dims/lane)
    for (int r = 0; r < 8; ++r) {
      int tkn = base + half * 8 + r;
      if (mp[tkn] > 0.0f) {
        int lab = labels[r];
        const __bf16* trow = tok + (size_t)tkn * DD + l15 * 48;
        float* dst = &lsum[lab * DD + l15 * 48];
#pragma unroll
        for (int j = 0; j < 48; j += 8) {
          v8bf tv = *(const v8bf*)(trow + j);
#pragma unroll
          for (int e = 0; e < 8; ++e) atomicAdd(&dst[j + e], (float)tv[e]);
        }
        if (l15 == 0) atomicAdd(&lcnt[lab], 1.0f);
      }
    }
  }
  __syncthreads();
  for (int i = threadIdx.x; i < CC * DD; i += 256) {
    int c = i / DD, d = i - c * DD;
    atomicAdd(&gsums[((size_t)k * CPAD + c) * DD + d], lsum[i]);
  }
  if (threadIdx.x < CC) atomicAdd(&gcnt[k * CPAD + threadIdx.x], lcnt[threadIdx.x]);
}

// ---------------------------------------------------------------------------
// Kernel 4: centroid update: c = counts>0 ? sums/max(counts,1) : c_old.
// Recompute ||c||^2 (f32 + bf16 copies), zero accumulators for next iter.
// ---------------------------------------------------------------------------
__global__ void __launch_bounds__(256) update_cent_k(
    float* __restrict__ centf, unsigned short* __restrict__ centb,
    float* __restrict__ cn2, float* __restrict__ gsums,
    float* __restrict__ gcnt) {
  int k = blockIdx.x / CC, c = blockIdx.x % CC;
  size_t row = ((size_t)k * CPAD + c) * DD;
  float cnt = gcnt[k * CPAD + c];
  __shared__ float red[256];
  float ss = 0.0f;
  for (int jj = 0; jj < 3; ++jj) {
    int d = threadIdx.x + jj * 256;
    float s = gsums[row + d];
    float v = (cnt > 0.0f) ? (s / fmaxf(cnt, 1.0f)) : centf[row + d];
    centf[row + d] = v;
    centb[row + d] = f2bf(v);
    ss += v * v;
    gsums[row + d] = 0.0f;  // ready for next iteration
  }
  red[threadIdx.x] = ss;
  __syncthreads();
  for (int s2 = 128; s2 > 0; s2 >>= 1) {
    if (threadIdx.x < s2) red[threadIdx.x] += red[threadIdx.x + s2];
    __syncthreads();
  }
  if (threadIdx.x == 0) {
    cn2[k * CPAD + c] = red[0];
    gcnt[k * CPAD + c] = 0.0f;
  }
}

// ---------------------------------------------------------------------------
// Kernel 5: tiny 80x768x768 MLP layer in exact f32 (precision-sensitive tail,
// <1% of runtime). One block per output row; input row cached in LDS.
// ---------------------------------------------------------------------------
__global__ void __launch_bounds__(256) mlp_k(
    const float* __restrict__ in, const float* __restrict__ Wm,
    const float* __restrict__ bb, float* __restrict__ out, int in_is_cent,
    int do_gelu) {
  __shared__ float lc[DD];
  int m = blockIdx.x;
  const float* src;
  if (in_is_cent) {
    int k = m / CC, c = m % CC;
    src = in + ((size_t)k * CPAD + c) * DD;
  } else {
    src = in + (size_t)m * DD;
  }
  for (int jj = 0; jj < 3; ++jj) {
    int d = threadIdx.x + jj * 256;
    lc[d] = src[d];
  }
  __syncthreads();
  for (int jj = 0; jj < 3; ++jj) {
    int j = threadIdx.x + jj * 256;
    float acc = bb[j];
#pragma unroll 4
    for (int d = 0; d < DD; ++d) acc += lc[d] * Wm[(size_t)d * DD + j];
    out[(size_t)m * DD + j] = do_gelu ? gelu_exact(acc) : acc;
  }
}

// ---------------------------------------------------------------------------
extern "C" void kernel_launch(void* const* d_in, const int* in_sizes, int n_in,
                              void* d_out, int out_size, void* d_ws,
                              size_t ws_size, hipStream_t stream) {
  (void)in_sizes; (void)n_in; (void)out_size; (void)ws_size;
  const float* feat = (const float*)d_in[0];
  const float* mask = (const float*)d_in[1];
  const float* boxes = (const float*)d_in[2];
  const float* Wp = (const float*)d_in[3];
  const float* bp = (const float*)d_in[4];
  const float* W1 = (const float*)d_in[5];
  const float* b1 = (const float*)d_in[6];
  const float* W2 = (const float*)d_in[7];
  const float* b2 = (const float*)d_in[8];
  const int* init_idx = (const int*)d_in[9];
  const int* rawh = (const int*)d_in[10];
  const int* raww = (const int*)d_in[11];
  float* out = (float*)d_out;

  char* ws = (char*)d_ws;
  size_t off = 0;
  auto carve = [&](size_t bytes) -> char* {
    char* p = ws + off;
    off += (bytes + 255) & ~(size_t)255;
    return p;
  };
  unsigned short* tokens = (unsigned short*)carve((size_t)KIMG * NN * DD * 2);
  float* centf = (float*)carve((size_t)KIMG * CPAD * DD * 4);
  unsigned short* centb = (unsigned short*)carve((size_t)KIMG * CPAD * DD * 2);
  float* cn2 = (float*)carve((size_t)KIMG * CPAD * 4);
  float* gsums = (float*)carve((size_t)KIMG * CPAD * DD * 4);
  float* gcnt = (float*)carve((size_t)KIMG * CPAD * 4);
  float* h1 = (float*)carve((size_t)KIMG * CC * DD * 4);

  build_tokens_k<<<KIMG * NN, DD / 4, 0, stream>>>(feat, mask, boxes, Wp, bp,
                                                   rawh, raww, tokens);
  init_cent_k<<<KIMG * CPAD, 256, 0, stream>>>(feat, boxes, Wp, bp, init_idx,
                                               rawh, raww, centf, centb, cn2,
                                               gsums, gcnt);
  for (int it = 0; it < KM_ITERS; ++it) {
    assign_k<<<KIMG * (NN / TPB), 256, 0, stream>>>(
        (const __bf16*)tokens, (const __bf16*)centb, cn2, mask, gsums, gcnt);
    update_cent_k<<<KIMG * CC, 256, 0, stream>>>(centf, centb, cn2, gsums,
                                                 gcnt);
  }
  mlp_k<<<KIMG * CC, 256, 0, stream>>>(centf, W1, b1, h1, 1, 1);
  mlp_k<<<KIMG * CC, 256, 0, stream>>>(h1, W2, b2, out, 0, 0);
}
